// LSDA_53352083750981
// MI455X (gfx1250) — compile-verified, matrix operand order checked by reading, not run on hardware
//
#include <hip/hip_runtime.h>
#include <hip/hip_bf16.h>

typedef __attribute__((ext_vector_type(16))) _Float16 v16h;
typedef __attribute__((ext_vector_type(8)))  float    v8f;

#define TWOH   512
#define HDIM   256
#define G3     768
#define KXPAD  320   // I=300 padded to multiple of 32

// ---------------------------------------------------------------------------
// f32 -> f16 convert with K padding (used for x and all GRU weights)
// ---------------------------------------------------------------------------
__global__ void k_cvt_pad(const float* __restrict__ src, _Float16* __restrict__ dst,
                          int rows, int srcK, int dstK) {
    size_t idx = (size_t)blockIdx.x * blockDim.x + threadIdx.x;
    size_t total = (size_t)rows * (size_t)dstK;
    if (idx >= total) return;
    int c = (int)(idx % dstK);
    size_t r = idx / dstK;
    dst[idx] = (c < srcK) ? (_Float16)src[r * (size_t)srcK + c] : (_Float16)0.0f;
}

// ---------------------------------------------------------------------------
// WMMA helpers (CDNA5 16x16x32 f16 -> f32), layouts per CDNA5 ISA
// A frag: lane<16 holds M=lane, K in {k0..k0+7, k0+16..k0+23};
//         lane>=16 holds M=lane-16, K in {k0+8..15, k0+24..31}
// ---------------------------------------------------------------------------
__device__ __forceinline__ v16h load_frag16(const _Float16* rowp, int k0, int khalf) {
    v16h v;
    const _Float16* p = rowp + k0 + khalf * 8;
#pragma unroll
    for (int j = 0; j < 8; ++j) v[j] = p[j];
#pragma unroll
    for (int j = 0; j < 8; ++j) v[8 + j] = p[16 + j];
    return v;
}

__device__ __forceinline__ v8f wmma16(v16h a, v16h b, v8f c) {
    return __builtin_amdgcn_wmma_f32_16x16x32_f16(false, a, false, b, (short)0, c,
                                                  false, false);
}

// ---------------------------------------------------------------------------
// One GRU direction. Each block owns a 16-row batch tile and iterates time
// sequentially; hidden state lives in LDS (f16). Per step:
//   xg = x_t @ Wih^T   (K=320, WMMA)     hg = h @ Whh^T  (K=256, WMMA)
//   r = sig(xr+hr) ; z = sig(xz+hz) ; n = tanh(xn + r*hn)
//   h = (1-z)*n + z*h ;  out[b,t,colofs:colofs+256] = h
// 256 threads = 8 waves; wave w owns N-tiles 6w..6w+5 of the 48 gate tiles.
// Weight fragments are deliberately re-fetched from global (L2-resident)
// every tile: an asm barrier on the *offset* (not the pointer, so addrspace
// inference keeps the loads as global_load, LOADcnt-only) defeats LICM so the
// compiler cannot hoist ~864 VGPRs of weight fragments and spill to scratch.
// ---------------------------------------------------------------------------
__global__ void __launch_bounds__(256)
k_gru_dir(const _Float16* __restrict__ xf16, int T,
          const _Float16* __restrict__ Wih, const _Float16* __restrict__ Whh,
          const float* __restrict__ bih, const float* __restrict__ bhh,
          float* __restrict__ out, int colofs, int reverse) {
    __shared__ _Float16 sh_h[16 * HDIM];   //  8 KB hidden state tile (f16)
    __shared__ float    sg [16 * G3];      // 48 KB: r/z = x+h sum; n cols = h part
    __shared__ float    sxn[16 * HDIM];    // 16 KB: x part of n gate

    const int tid   = threadIdx.x;
    const int lane  = tid & 31;
    const int wave  = tid >> 5;
    const int b0    = blockIdx.x * 16;
    const int mrow  = lane & 15;
    const int khalf = lane >> 4;

    for (int i = tid; i < 16 * HDIM; i += 256) sh_h[i] = (_Float16)0.0f;
    __syncthreads();

    for (int s = 0; s < T; ++s) {
        const int t = reverse ? (T - 1 - s) : s;
        const _Float16* xrow = xf16 + ((size_t)(b0 + mrow) * T + t) * KXPAD;
        const _Float16* hrow = &sh_h[mrow * HDIM];
        if (s + 1 < T) {   // hint next timestep's activations into cache
            int tn = reverse ? (T - 2 - s) : (s + 1);
            __builtin_prefetch(xf16 + ((size_t)(b0 + mrow) * T + tn) * KXPAD, 0, 1);
        }
#pragma unroll 1
        for (int q = 0; q < 6; ++q) {
            const int nt   = wave * 6 + q;
            const int ncol = nt * 16;
            // Opaque offsets: loads cannot be hoisted out of the time loop
            // (no spills), yet the base pointers keep global address space
            // so these lower to global_load_b128 (LOADcnt-only tracking).
            size_t wofs = (size_t)(ncol + mrow) * KXPAD;
            size_t hofs = (size_t)(ncol + mrow) * HDIM;
            asm volatile("" : "+v"(wofs), "+v"(hofs));
            const _Float16* wihrow = Wih + wofs;
            const _Float16* whhrow = Whh + hofs;
            v8f accx = {};
            v8f acch = {};
#pragma unroll
            for (int kk = 0; kk < KXPAD / 32; ++kk) {
                v16h a = load_frag16(xrow,   kk * 32, khalf);
                v16h b = load_frag16(wihrow, kk * 32, khalf);
                accx = wmma16(a, b, accx);
            }
#pragma unroll
            for (int kk = 0; kk < HDIM / 32; ++kk) {
                v16h a = load_frag16(hrow,   kk * 32, khalf);
                v16h b = load_frag16(whhrow, kk * 32, khalf);
                acch = wmma16(a, b, acch);
            }
            // D tile: VGPR r -> M = r + 8*(lane>=16), N = lane&15
            const int mb = khalf * 8;
            if (nt < 32) {          // r and z gate ranges: store x+h sum
#pragma unroll
                for (int r = 0; r < 8; ++r)
                    sg[(mb + r) * G3 + ncol + mrow] = accx[r] + acch[r];
            } else {                // n gate range: keep x and h parts apart
#pragma unroll
                for (int r = 0; r < 8; ++r) {
                    sg [(mb + r) * G3   + ncol + mrow]         = acch[r];
                    sxn[(mb + r) * HDIM + (ncol - 512) + mrow] = accx[r];
                }
            }
        }
        __syncthreads();
        // elementwise GRU update: 16 m-rows x 256 cols, 16 elems per thread
        {
            const int m  = tid >> 4;
            const int j0 = (tid & 15) * 16;
#pragma unroll
            for (int jj = 0; jj < 16; ++jj) {
                const int j = j0 + jj;
                float xr_hr = sg[m * G3 + j]       + bih[j]       + bhh[j];
                float xz_hz = sg[m * G3 + 256 + j] + bih[256 + j] + bhh[256 + j];
                float hn    = sg[m * G3 + 512 + j] + bhh[512 + j];
                float xn    = sxn[m * HDIM + j]    + bih[512 + j];
                float r = 1.0f / (1.0f + __expf(-xr_hr));
                float z = 1.0f / (1.0f + __expf(-xz_hz));
                float n = tanhf(xn + r * hn);
                float ho   = (float)sh_h[m * HDIM + j];
                float hnew = (1.0f - z) * n + z * ho;
                sh_h[m * HDIM + j] = (_Float16)hnew;
                out[((size_t)(b0 + m) * T + t) * TWOH + colofs + j] = hnew;
            }
        }
        __syncthreads();
    }
}

// ---------------------------------------------------------------------------
// Similarity-aware attention pooling: u=tanh(out@W+b), softmax over T,
// L = sum_t a_t * out_t.  One block per batch row, 256 threads (8 waves).
// ---------------------------------------------------------------------------
__global__ void __launch_bounds__(256)
k_attn(const float* __restrict__ out, int T,
       const float* __restrict__ attW, const float* __restrict__ attb,
       float* __restrict__ L) {
    __shared__ float su[512];
    __shared__ float red[256];
    __shared__ float ws[TWOH];
    const int b = blockIdx.x, tid = threadIdx.x;
    const int ln = tid & 31, wv = tid >> 5;
    for (int c = tid; c < TWOH; c += 256) ws[c] = attW[c];
    __syncthreads();
    const float ab = attb[0];
    const float* ob = out + (size_t)b * T * TWOH;
    for (int t0 = 0; t0 < T; t0 += 8) {     // 8 waves each score one t
        const int t = t0 + wv;
        const float* row = ob + (size_t)t * TWOH;
        float p = 0.0f;
        for (int c = ln; c < TWOH; c += 32) p += row[c] * ws[c];
#pragma unroll
        for (int o = 16; o > 0; o >>= 1) p += __shfl_xor(p, o, 32);
        if (ln == 0) su[t] = tanhf(p + ab);
    }
    __syncthreads();
    float m = -3.4e38f;
    for (int t = tid; t < T; t += 256) m = fmaxf(m, su[t]);
    red[tid] = m; __syncthreads();
    for (int s = 128; s > 0; s >>= 1) {
        if (tid < s) red[tid] = fmaxf(red[tid], red[tid + s]);
        __syncthreads();
    }
    const float mx = red[0]; __syncthreads();
    float lsum = 0.0f;
    for (int t = tid; t < T; t += 256) { float e = __expf(su[t] - mx); su[t] = e; lsum += e; }
    red[tid] = lsum; __syncthreads();
    for (int s = 128; s > 0; s >>= 1) {
        if (tid < s) red[tid] += red[tid + s];
        __syncthreads();
    }
    const float inv = 1.0f / red[0];
    __syncthreads();
    float a0 = 0.0f, a1 = 0.0f;
    for (int t = 0; t < T; ++t) {
        const float a = su[t] * inv;
        const float* row = ob + (size_t)t * TWOH;
        a0 += a * row[tid];
        a1 += a * row[tid + 256];
    }
    L[(size_t)b * TWOH + tid]       = a0;
    L[(size_t)b * TWOH + tid + 256] = a1;
}

// ---------------------------------------------------------------------------
// Tail: cos blocks, mlp, att1/att2 rank-structured attention, mlp2.
// hbar2[d][j] = sum_i tanh(W1[i]*Lc[j]) * W2[i][d]; global softmax; bmm; mlp2.
// One block per batch row, 64 threads.
// ---------------------------------------------------------------------------
__global__ void __launch_bounds__(64)
k_final(const float* __restrict__ Lh, const float* __restrict__ Lb,
        const float* __restrict__ mlpW, const float* __restrict__ mlpb,
        const float* __restrict__ W1,  const float* __restrict__ W2,
        const float* __restrict__ m2W, const float* __restrict__ m2b,
        float* __restrict__ P) {
    __shared__ float lh[TWOH], lb[TWOH];
    __shared__ float Lc[48];
    __shared__ float tt[48 * 48];
    __shared__ float h2[48 * 48];
    __shared__ float red[64];
    const int b = blockIdx.x, tid = threadIdx.x;
    for (int i = tid; i < TWOH; i += 64) {
        lh[i] = Lh[(size_t)b * TWOH + i];
        lb[i] = Lb[(size_t)b * TWOH + i];
    }
    __syncthreads();
    if (tid < 16) {                                   // cosine similarity blocks
        float d = 0, na = 0, nb = 0;
        for (int j = 0; j < 32; ++j) {
            float x = lh[tid * 32 + j], y = lb[tid * 32 + j];
            d += x * y; na += x * x; nb += y * y;
        }
        Lc[16 + tid] = d / (fmaxf(sqrtf(na), 1e-8f) * fmaxf(sqrtf(nb), 1e-8f));
    } else if (tid < 32) {                            // Lhk
        int k = tid - 16;
        float s = 0; const float* w = mlpW + (size_t)k * TWOH;
        for (int j = 0; j < TWOH; ++j) s += lh[j] * w[j];
        Lc[k] = s + mlpb[k];
    } else if (tid < 48) {                            // Lbk
        int k = tid - 32;
        float s = 0; const float* w = mlpW + (size_t)k * TWOH;
        for (int j = 0; j < TWOH; ++j) s += lb[j] * w[j];
        Lc[32 + k] = s + mlpb[k];
    }
    __syncthreads();
    for (int idx = tid; idx < 2304; idx += 64) {      // tanh(W1[i]*Lc[j]) table
        int i = idx / 48, j = idx % 48;
        tt[idx] = tanhf(W1[i] * Lc[j]);
    }
    __syncthreads();
    for (int idx = tid; idx < 2304; idx += 64) {      // hbar2[d][j]
        int d = idx / 48, j = idx % 48;
        float s = 0.0f;
        for (int i = 0; i < 48; ++i) s += tt[i * 48 + j] * W2[i * 48 + d];
        h2[idx] = s;
    }
    __syncthreads();
    float m = -3.4e38f;                               // global softmax over 2304
    for (int idx = tid; idx < 2304; idx += 64) m = fmaxf(m, h2[idx]);
    red[tid] = m; __syncthreads();
    for (int s = 32; s > 0; s >>= 1) {
        if (tid < s) red[tid] = fmaxf(red[tid], red[tid + s]);
        __syncthreads();
    }
    const float mx = red[0]; __syncthreads();
    float lsum = 0.0f;
    for (int idx = tid; idx < 2304; idx += 64) {
        float e = __expf(h2[idx] - mx); h2[idx] = e; lsum += e;
    }
    red[tid] = lsum; __syncthreads();
    for (int s = 32; s > 0; s >>= 1) {
        if (tid < s) red[tid] += red[tid + s];
        __syncthreads();
    }
    const float inv = 1.0f / red[0];
    __syncthreads();
    if (tid < 48) {                                   // A_prod[d] = sum_j A*Lc
        float s = 0.0f;
        for (int j = 0; j < 48; ++j) s += h2[tid * 48 + j] * Lc[j];
        red[tid] = s * inv;
    }
    __syncthreads();
    if (tid < 2) {                                    // P = A_prod @ mlp2W^T + b
        float s = 0.0f;
        for (int d = 0; d < 48; ++d) s += red[d] * m2W[tid * 48 + d];
        P[(size_t)b * 2 + tid] = s + m2b[tid];
    }
}

// ---------------------------------------------------------------------------
// Host launcher. Input order (flattened setup_inputs dict):
//  0 head  1 body
//  gru_h: 2 Wih_f 3 Whh_f 4 bih_f 5 bhh_f 6 Wih_b 7 Whh_b 8 bih_b 9 bhh_b
//  gru_b: 10..17 (same order)
//  18 att_h_W 19 att_h_b 20 att_b_W 21 att_b_b 22 mlp_W 23 mlp_b
//  24 att1_W 25 att2_W 26 mlp2_W 27 mlp2_b
// Output: P(256*2) | Lh(256*512) | Lb(256*512)
// ---------------------------------------------------------------------------
extern "C" void kernel_launch(void* const* d_in, const int* in_sizes, int n_in,
                              void* d_out, int out_size, void* d_ws, size_t ws_size,
                              hipStream_t stream) {
    (void)in_sizes; (void)n_in; (void)out_size; (void)ws_size;
    const float* head = (const float*)d_in[0];
    const float* body = (const float*)d_in[1];

    char* base = (char*)d_ws;
    size_t off = 0;
    auto take = [&](size_t bytes) -> char* {
        char* p = base + off;
        off = (off + bytes + 255) & ~(size_t)255;
        return p;
    };
    _Float16* xh     = (_Float16*)take((size_t)8192   * KXPAD * 2);
    _Float16* xb     = (_Float16*)take((size_t)131072 * KXPAD * 2);
    _Float16* wih_hf = (_Float16*)take((size_t)G3 * KXPAD * 2);
    _Float16* wih_hb = (_Float16*)take((size_t)G3 * KXPAD * 2);
    _Float16* wih_bf = (_Float16*)take((size_t)G3 * KXPAD * 2);
    _Float16* wih_bb = (_Float16*)take((size_t)G3 * KXPAD * 2);
    _Float16* whh_hf = (_Float16*)take((size_t)G3 * HDIM * 2);
    _Float16* whh_hb = (_Float16*)take((size_t)G3 * HDIM * 2);
    _Float16* whh_bf = (_Float16*)take((size_t)G3 * HDIM * 2);
    _Float16* whh_bb = (_Float16*)take((size_t)G3 * HDIM * 2);
    float*    out_h  = (float*)take((size_t)8192   * TWOH * 4);
    float*    out_b  = (float*)take((size_t)131072 * TWOH * 4);

    auto cvt = [&](const void* src, _Float16* dst, int rows, int sk, int dk) {
        size_t tot = (size_t)rows * dk;
        int blocks = (int)((tot + 255) / 256);
        k_cvt_pad<<<blocks, 256, 0, stream>>>((const float*)src, dst, rows, sk, dk);
    };
    cvt(head,     xh,     8192,   300, KXPAD);
    cvt(body,     xb,     131072, 300, KXPAD);
    cvt(d_in[2],  wih_hf, G3, 300, KXPAD);
    cvt(d_in[6],  wih_hb, G3, 300, KXPAD);
    cvt(d_in[10], wih_bf, G3, 300, KXPAD);
    cvt(d_in[14], wih_bb, G3, 300, KXPAD);
    cvt(d_in[3],  whh_hf, G3, 256, HDIM);
    cvt(d_in[7],  whh_hb, G3, 256, HDIM);
    cvt(d_in[11], whh_bf, G3, 256, HDIM);
    cvt(d_in[15], whh_bb, G3, 256, HDIM);

    // head fwd / bwd, body fwd / bwd (batch-parallel recurrences, WMMA GEMMs)
    k_gru_dir<<<16, 256, 0, stream>>>(xh, 32,  wih_hf, whh_hf,
        (const float*)d_in[4],  (const float*)d_in[5],  out_h, 0,   0);
    k_gru_dir<<<16, 256, 0, stream>>>(xh, 32,  wih_hb, whh_hb,
        (const float*)d_in[8],  (const float*)d_in[9],  out_h, 256, 1);
    k_gru_dir<<<16, 256, 0, stream>>>(xb, 512, wih_bf, whh_bf,
        (const float*)d_in[12], (const float*)d_in[13], out_b, 0,   0);
    k_gru_dir<<<16, 256, 0, stream>>>(xb, 512, wih_bb, whh_bb,
        (const float*)d_in[16], (const float*)d_in[17], out_b, 256, 1);

    float* P   = (float*)d_out;
    float* Lh  = P + 256 * 2;
    float* Lb  = Lh + 256 * TWOH;
    k_attn<<<256, 256, 0, stream>>>(out_h, 32,  (const float*)d_in[18],
                                    (const float*)d_in[19], Lh);
    k_attn<<<256, 256, 0, stream>>>(out_b, 512, (const float*)d_in[20],
                                    (const float*)d_in[21], Lb);
    k_final<<<256, 64, 0, stream>>>(Lh, Lb,
        (const float*)d_in[22], (const float*)d_in[23],
        (const float*)d_in[24], (const float*)d_in[25],
        (const float*)d_in[26], (const float*)d_in[27], P);
}
